// Attention_15006615734250
// MI455X (gfx1250) — compile-verified, matrix operand order checked by reading, not run on hardware
//
#include <hip/hip_runtime.h>

// Problem constants (B=4, N=2048, C=768, H=6, hd=128)
#define BB   4
#define NN   2048
#define CC   768
#define HD   128
#define SCALE 0.08838834764831845f   // 1/sqrt(128)

typedef __attribute__((ext_vector_type(16))) __bf16 v16bf;
typedef __attribute__((ext_vector_type(8)))  float  v8f;

union FragU { uint4 u[2]; v16bf v; };

// Load a 16x32 bf16 fragment (A-operand layout; identical per-lane packing is
// used for the B operand, which therefore consumes rows of B^T, i.e. we compute
// C = A @ Bm^T when Bm is stored row-major [N,K]).
// Per ISA 7.12.2: lanes 0-15 hold K={0..7,16..23}, lanes 16-31 hold K={8..15,24..31}.
__device__ __forceinline__ v16bf ldfrag(const __bf16* p0, int ldh) {
  const int lane = threadIdx.x & 31;
  const int r  = lane & 15;
  const int ko = (lane & 16) ? 8 : 0;
  const __bf16* p = p0 + (size_t)r * ldh + ko;
  FragU f;
  f.u[0] = *(const uint4*)(p);
  f.u[1] = *(const uint4*)(p + 16);
  return f.v;
}

__device__ __forceinline__ v8f wmma_bf16(v16bf a, v16bf b, v8f c) {
  return __builtin_amdgcn_wmma_f32_16x16x32_bf16(false, a, false, b, (short)0, c,
                                                 false, false);
}

// ---------------------------------------------------------------- fp32 -> bf16
__global__ void __launch_bounds__(256)
cvt_bf16_kernel(const float* __restrict__ src, __bf16* __restrict__ dst, size_t n) {
  size_t i = (size_t)blockIdx.x * blockDim.x + threadIdx.x;
  if (i < n) dst[i] = (__bf16)src[i];
}

// ------------------------------------------------------- generic bf16 GEMM
// C[M,N] = A[M,K] @ Bm[N,K]^T ; wave computes a 16x128 strip (8 WMMA tiles).
// grid = (N/128, M/128), block = 256 (8 waves).
template <bool BF16OUT, bool BIAS>
__global__ void __launch_bounds__(256)
gemm_kernel(const __bf16* __restrict__ A,
            const __bf16* __restrict__ Bm,
            void* __restrict__ Cout,
            const float* __restrict__ bias,
            int ksteps, int lda, int ldb, int ldc) {
  const int lane = threadIdx.x & 31;
  const int w    = threadIdx.x >> 5;
  const int r0   = blockIdx.y * 128 + w * 16;
  const int c0   = blockIdx.x * 128;
  const __bf16* Ab = A + (size_t)r0 * lda;

  v8f zero = {0.f,0.f,0.f,0.f,0.f,0.f,0.f,0.f};
  v8f acc[8];
#pragma unroll
  for (int nt = 0; nt < 8; nt++) acc[nt] = zero;

  for (int kk = 0; kk < ksteps; kk++) {
    v16bf af = ldfrag(Ab + kk * 32, lda);
#pragma unroll
    for (int nt = 0; nt < 8; nt++) {
      v16bf bf = ldfrag(Bm + (size_t)(c0 + nt * 16) * ldb + kk * 32, ldb);
      acc[nt] = wmma_bf16(af, bf, acc[nt]);
    }
  }
  const int halfoff = (lane & 16) >> 1;   // C layout: lanes>=16 hold rows M+8
#pragma unroll
  for (int nt = 0; nt < 8; nt++) {
#pragma unroll
    for (int e = 0; e < 8; e++) {
      int row = r0 + e + halfoff;
      int col = c0 + nt * 16 + (lane & 15);
      if (BF16OUT) {
        ((__bf16*)Cout)[(size_t)row * ldc + col] = (__bf16)acc[nt][e];
      } else {
        float v = acc[nt][e];
        if (BIAS) v += bias[col];
        ((float*)Cout)[(size_t)row * ldc + col] = v;
      }
    }
  }
}

// ------------------------------------------- V (pose heads) -> Vt [bh][d][n]
__global__ void __launch_bounds__(256)
transpose_v_kernel(const __bf16* __restrict__ qkvb, __bf16* __restrict__ vt) {
  size_t idx = (size_t)blockIdx.x * 256 + threadIdx.x;  // 12*128*2048 total
  int n  = (int)(idx & 2047);
  int d  = (int)((idx >> 11) & 127);
  int bh = (int)(idx >> 18);
  int b = bh / 3, hp = bh % 3;
  vt[idx] = qkvb[((size_t)(b * NN + n)) * (3 * CC) + 2 * CC + (3 + hp) * HD + d];
}

// --------------------------------- sigmoid attention + ctx1 = attn_obj @ Vpose
// grid = (N/128, B*3); each wave owns 16 q-rows. Writes fp32 attn_obj to d_out
// and ctx1^T (bf16, [bh][d][q]) for the pose stage.
// kk-outer loop keeps only ONE Q/K fragment live at a time (no spills).
__global__ void __launch_bounds__(256)
attn_obj_kernel(const __bf16* __restrict__ qkvb,
                const __bf16* __restrict__ vt,
                float* __restrict__ attn,
                __bf16* __restrict__ ctx1t) {
  __shared__ __align__(16) __bf16 P_lds[128 * 136];
  const int lane = threadIdx.x & 31;
  const int w    = threadIdx.x >> 5;
  const int bh   = blockIdx.y;
  const int b    = bh / 3, h = bh % 3;
  const int q0   = blockIdx.x * 128 + w * 16;
  const int halfoff = (lane & 16) >> 1;

  const __bf16* Qbase = qkvb + ((size_t)(b * NN + q0)) * (3 * CC) + h * HD;
  const __bf16* Kbase = qkvb + ((size_t)(b * NN)) * (3 * CC) + CC + h * HD;
  const __bf16* Vtb   = vt + (size_t)bh * HD * NN;

  v8f zero = {0.f,0.f,0.f,0.f,0.f,0.f,0.f,0.f};
  v8f oacc[8];
#pragma unroll
  for (int nt = 0; nt < 8; nt++) oacc[nt] = zero;

  for (int kb = 0; kb < 16; kb++) {
    const int kbase = kb * 128;
    // S = Q K^T  (kk-outer: one qf live)
    v8f s[8];
#pragma unroll
    for (int nt = 0; nt < 8; nt++) s[nt] = zero;
#pragma unroll
    for (int kk = 0; kk < 4; kk++) {
      v16bf qf = ldfrag(Qbase + kk * 32, 3 * CC);
#pragma unroll
      for (int nt = 0; nt < 8; nt++) {
        v16bf kf = ldfrag(Kbase + (size_t)(kbase + nt * 16) * (3 * CC) + kk * 32,
                          3 * CC);
        s[nt] = wmma_bf16(qf, kf, s[nt]);
      }
    }
    // sigmoid -> attn_obj (fp32) + P_lds (bf16, wave-private rows)
#pragma unroll
    for (int nt = 0; nt < 8; nt++) {
#pragma unroll
      for (int e = 0; e < 8; e++) {
        float p = 1.f / (1.f + __expf(-(s[nt][e] * SCALE)));
        int ml = e + halfoff;
        int q  = q0 + ml;
        int kc = kbase + nt * 16 + (lane & 15);
        attn[((size_t)bh * NN + q) * NN + kc] = p;
        P_lds[(w * 16 + ml) * 136 + nt * 16 + (lane & 15)] = (__bf16)p;
      }
    }
    // ctx1 += P @ Vpose (B operand = rows of V^T = Vt rows)
#pragma unroll
    for (int kk = 0; kk < 4; kk++) {
      v16bf pf = ldfrag(P_lds + w * 16 * 136 + kk * 32, 136);
#pragma unroll
      for (int nt = 0; nt < 8; nt++) {
        v16bf vf = ldfrag(Vtb + (size_t)(nt * 16) * NN + kbase + kk * 32, NN);
        oacc[nt] = wmma_bf16(pf, vf, oacc[nt]);
      }
    }
  }
  // store ctx1^T : ctx1t[bh][d][q], 8 consecutive q per lane -> packed b128
#pragma unroll
  for (int nt = 0; nt < 8; nt++) {
    int d = nt * 16 + (lane & 15);
    union { __bf16 h8[8]; uint4 u; } pk;
#pragma unroll
    for (int e = 0; e < 8; e++) pk.h8[e] = (__bf16)oacc[nt][e];
    *(uint4*)(ctx1t + ((size_t)bh * HD + d) * NN + q0 + halfoff) = pk.u;
  }
}

// --------------------------- flash softmax: ctx2 = softmax(QpKp^T*s) @ ctx1
// grid = (N/128, B*3). Writes ctx2 bf16 into ctxb[b][q][hp*128+d].
__global__ void __launch_bounds__(256)
attn_pose_kernel(const __bf16* __restrict__ qkvb,
                 const __bf16* __restrict__ ctx1t,
                 __bf16* __restrict__ ctxb) {
  __shared__ __align__(16) __bf16 P_lds[128 * 136];
  const int lane = threadIdx.x & 31;
  const int w    = threadIdx.x >> 5;
  const int bh   = blockIdx.y;
  const int b    = bh / 3, hp = bh % 3;
  const int q0   = blockIdx.x * 128 + w * 16;
  const int halfoff = (lane & 16) >> 1;

  const __bf16* Qbase = qkvb + ((size_t)(b * NN + q0)) * (3 * CC) + (3 + hp) * HD;
  const __bf16* Kbase = qkvb + ((size_t)(b * NN)) * (3 * CC) + CC + (3 + hp) * HD;
  const __bf16* Vtb   = ctx1t + (size_t)bh * HD * NN;

  v8f zero = {0.f,0.f,0.f,0.f,0.f,0.f,0.f,0.f};
  v8f oacc[8];
  float mi[8], li[8];
#pragma unroll
  for (int nt = 0; nt < 8; nt++) oacc[nt] = zero;
#pragma unroll
  for (int e = 0; e < 8; e++) { mi[e] = -1e30f; li[e] = 0.f; }

  for (int kb = 0; kb < 16; kb++) {
    const int kbase = kb * 128;
    // S = Q K^T  (kk-outer: one qf live)
    v8f s[8];
#pragma unroll
    for (int nt = 0; nt < 8; nt++) s[nt] = zero;
#pragma unroll
    for (int kk = 0; kk < 4; kk++) {
      v16bf qf = ldfrag(Qbase + kk * 32, 3 * CC);
#pragma unroll
      for (int nt = 0; nt < 8; nt++) {
        v16bf kf = ldfrag(Kbase + (size_t)(kbase + nt * 16) * (3 * CC) + kk * 32,
                          3 * CC);
        s[nt] = wmma_bf16(qf, kf, s[nt]);
      }
    }
    // online softmax row stats (rows are per (e, lane-half); reduce over 16 lanes)
    float corr[8];
#pragma unroll
    for (int e = 0; e < 8; e++) {
      float tm = -1e30f;
#pragma unroll
      for (int nt = 0; nt < 8; nt++) {
        s[nt][e] *= SCALE;
        tm = fmaxf(tm, s[nt][e]);
      }
#pragma unroll
      for (int off = 1; off < 16; off <<= 1)
        tm = fmaxf(tm, __shfl_xor(tm, off, 16));
      float nm = fmaxf(mi[e], tm);
      corr[e] = __expf(mi[e] - nm);
      mi[e] = nm;
    }
    // P = exp(s - m), stage into LDS (wave-private rows), rowsum
#pragma unroll
    for (int nt = 0; nt < 8; nt++) {
#pragma unroll
      for (int e = 0; e < 8; e++) {
        float p = __expf(s[nt][e] - mi[e]);
        s[nt][e] = p;
        int ml = e + halfoff;
        P_lds[(w * 16 + ml) * 136 + nt * 16 + (lane & 15)] = (__bf16)p;
      }
    }
#pragma unroll
    for (int e = 0; e < 8; e++) {
      float rs = 0.f;
#pragma unroll
      for (int nt = 0; nt < 8; nt++) rs += s[nt][e];
#pragma unroll
      for (int off = 1; off < 16; off <<= 1) rs += __shfl_xor(rs, off, 16);
      li[e] = li[e] * corr[e] + rs;
#pragma unroll
      for (int nt = 0; nt < 8; nt++) oacc[nt][e] *= corr[e];
    }
    // O += P @ ctx1 (B operand = rows of ctx1^T)
#pragma unroll
    for (int kk = 0; kk < 4; kk++) {
      v16bf pf = ldfrag(P_lds + w * 16 * 136 + kk * 32, 136);
#pragma unroll
      for (int nt = 0; nt < 8; nt++) {
        v16bf vf = ldfrag(Vtb + (size_t)(nt * 16) * NN + kbase + kk * 32, NN);
        oacc[nt] = wmma_bf16(pf, vf, oacc[nt]);
      }
    }
  }
  // normalize + store ctx2 into ctxb[b][q][hp*128 + d]
#pragma unroll
  for (int nt = 0; nt < 8; nt++) {
#pragma unroll
    for (int e = 0; e < 8; e++) {
      int q   = q0 + e + halfoff;
      int col = hp * HD + nt * 16 + (lane & 15);
      ctxb[((size_t)b * NN + q) * (CC / 2) + col] = (__bf16)(oacc[nt][e] / li[e]);
    }
  }
}

// ---------------------------------------------------------------- launcher
extern "C" void kernel_launch(void* const* d_in, const int* in_sizes, int n_in,
                              void* d_out, int out_size, void* d_ws, size_t ws_size,
                              hipStream_t stream) {
  (void)in_sizes; (void)n_in; (void)out_size; (void)ws_size;
  const float* x      = (const float*)d_in[0];   // [4,2048,768]
  const float* w_qkv  = (const float*)d_in[1];   // [2304,768]
  const float* w_proj = (const float*)d_in[2];   // [768,384]
  const float* b_proj = (const float*)d_in[3];   // [768]

  float* out  = (float*)d_out;                            // [4,2048,768]
  float* attn = out + (size_t)BB * NN * CC;               // [4,3,2048,2048]

  // workspace layout (bf16)
  __bf16* Xb     = (__bf16*)d_ws;                         // 8192 x 768
  __bf16* Wqkvb  = Xb     + (size_t)8192 * 768;           // 2304 x 768
  __bf16* Wprojb = Wqkvb  + (size_t)2304 * 768;           // 768  x 384
  __bf16* qkvb   = Wprojb + (size_t)768 * 384;            // 8192 x 2304
  __bf16* Vt     = qkvb   + (size_t)8192 * 2304;          // 12 x 128 x 2048
  __bf16* Ctx1t  = Vt     + (size_t)12 * 128 * 2048;      // 12 x 128 x 2048
  __bf16* Ctxb   = Ctx1t  + (size_t)12 * 128 * 2048;      // 8192 x 384

  cvt_bf16_kernel<<<(8192 * 768) / 256, 256, 0, stream>>>(x, Xb, (size_t)8192 * 768);
  cvt_bf16_kernel<<<(2304 * 768) / 256, 256, 0, stream>>>(w_qkv, Wqkvb, (size_t)2304 * 768);
  cvt_bf16_kernel<<<(768 * 384) / 256, 256, 0, stream>>>(w_proj, Wprojb, (size_t)768 * 384);

  // qkv = X @ Wqkv^T  (M=8192, N=2304, K=768)
  gemm_kernel<true, false><<<dim3(2304 / 128, 8192 / 128), 256, 0, stream>>>(
      Xb, Wqkvb, qkvb, nullptr, 768 / 32, 768, 768, 2304);

  transpose_v_kernel<<<(12 * 128 * 2048) / 256, 256, 0, stream>>>(qkvb, Vt);

  attn_obj_kernel<<<dim3(NN / 128, BB * 3), 256, 0, stream>>>(qkvb, Vt, attn, Ctx1t);
  attn_pose_kernel<<<dim3(NN / 128, BB * 3), 256, 0, stream>>>(qkvb, Ctx1t, Ctxb);

  // out = ctx @ Wproj^T + b  (M=8192, N=768, K=384)
  gemm_kernel<false, true><<<dim3(768 / 128, 8192 / 128), 256, 0, stream>>>(
      Ctxb, Wprojb, out, b_proj, 384 / 32, 384, 384, 768);
}
